// PointerGeneratorOut_66571993088449
// MI455X (gfx1250) — matrix operand order repaired and checked by reading, not compiled, not run on hardware
//
#include <hip/hip_runtime.h>
#include <hip/hip_bf16.h>

// ---------------------------------------------------------------------------
// Types for CDNA5 WMMA
// ---------------------------------------------------------------------------
typedef __attribute__((ext_vector_type(16))) __bf16        v16bf;
typedef __attribute__((ext_vector_type(8)))  float         v8f;
typedef __attribute__((ext_vector_type(4)))  unsigned int  v4u;

union FragU {                 // 32B fragment = two 16B LDS loads (ds_load_b128)
    v4u   q[2];
    v16bf v;
};

// 4-byte-aligned float pair (W rows with odd N are only dword aligned)
struct __attribute__((aligned(4))) f2a { float x, y; };

// pack two f32 into a dword of two bf16 (truncation; elem k -> low half)
static __device__ __forceinline__ unsigned int pack2bf(float lo, float hi) {
    unsigned int ulo = __builtin_bit_cast(unsigned int, lo) >> 16;
    unsigned int uhi = __builtin_bit_cast(unsigned int, hi) & 0xFFFF0000u;
    return uhi | ulo;
}

// ---------------------------------------------------------------------------
// Tiled bf16 WMMA GEMM:  out[M,N] = epilogue( X[M,K] @ W[K,N] + bias[N] )
// MODE 0: identity epilogue (logits)   MODE 1: tanh epilogue (gate hidden)
//
// Block = 256 threads = 8 waves; block tile 64x128; waves 4(m) x 2(n);
// each wave: 16x64 = 4 v_wmma_f32_16x16x32_bf16 per K-chunk (KC=32).
//
// LDS is double-buffered and fragment-native:
//   As: row-major [64][40] ushort (80B stride). A-fragment for lane
//       (m,half) = b128 at row*80 + half*32 and +16.
//   Bt: B transposed [128][40] ushort. B-fragment for lane (n,khalf) =
//       b128 at col*80 + khalf*32 and +16.
// Pipeline (one barrier/iter): load k+1 -> compute k -> store k+1 -> barrier.
// ---------------------------------------------------------------------------
#define KC 32
#define BM 64
#define BN 128
#define LDA 40   // ushorts per A row  (80 bytes)
#define LDB 40   // ushorts per Bt col (80 bytes)
#define ASZ (BM * LDA)
#define BSZ (BN * LDB)

template <int MODE, bool EDGE>
static __device__ __forceinline__ void pg_gemm_body(
    unsigned short* AsB, unsigned short* BtB,
    const float* __restrict__ X, const float* __restrict__ W,
    const float* __restrict__ bias, float* __restrict__ out,
    int N, int K, int block_m, int block_n)
{
    const int tid  = threadIdx.x;
    const int lane = tid & 31;
    const int wave = tid >> 5;
    const int wm   = wave & 3;   // 4 row tiles of 16  -> 64 rows
    const int wn   = wave >> 2;  // 2 col groups of 64 -> 128 cols

    // ---- loop-invariant staging geometry -------------------------------
    const float* aptr[4]; int a_off[4];
    const float* bptr[4]; int b_off0[4], b_off1[4];
    bool bv0[4], bv1[4];
    #pragma unroll
    for (int it = 0; it < 4; ++it) {
        int j    = tid + 256 * it;        // [0,1024)
        int arow = j >> 4, akp = j & 15;  // A: row 0..63, bf16-pair 0..15
        aptr[it]  = X + (size_t)(block_m + arow) * K + 2 * akp;
        a_off[it] = arow * LDA + akp * 2;
        int cp = j & 63, kp = j >> 6;     // B: col-pair 0..63, k-pair 0..15
        int gc = block_n + 2 * cp;
        bptr[it]   = W + (size_t)(2 * kp) * N + gc;
        b_off0[it] = (2 * cp)     * LDB + kp * 2;
        b_off1[it] = (2 * cp + 1) * LDB + kp * 2;
        bv0[it] = !EDGE || (gc < N);
        bv1[it] = !EDGE || (gc + 1 < N);
    }

    // staging registers (raw floats; packing deferred to store phase so the
    // global-load wait lands after the WMMAs)
    float2 fa[4];
    f2a    flo[4], fhi[4];

    auto stage_load = [&]() {
        #pragma unroll
        for (int it = 0; it < 4; ++it) {
            fa[it] = *(const float2*)aptr[it];
            aptr[it] += KC;
        }
        #pragma unroll
        for (int it = 0; it < 4; ++it) {
            if (!EDGE) {
                flo[it] = *(const f2a*)bptr[it];
                fhi[it] = *(const f2a*)(bptr[it] + N);
            } else {
                f2a z; z.x = 0.f; z.y = 0.f;
                flo[it] = z; fhi[it] = z;
                if (bv1[it]) {
                    flo[it] = *(const f2a*)bptr[it];
                    fhi[it] = *(const f2a*)(bptr[it] + N);
                } else if (bv0[it]) {
                    flo[it].x = bptr[it][0];
                    fhi[it].x = bptr[it][N];
                }
            }
            bptr[it] += (size_t)KC * N;
        }
    };
    auto stage_store = [&](unsigned short* As, unsigned short* Bt) {
        #pragma unroll
        for (int it = 0; it < 4; ++it)
            *(unsigned int*)(&As[a_off[it]]) = pack2bf(fa[it].x, fa[it].y);
        #pragma unroll
        for (int it = 0; it < 4; ++it) {
            *(unsigned int*)(&Bt[b_off0[it]]) = pack2bf(flo[it].x, fhi[it].x);
            *(unsigned int*)(&Bt[b_off1[it]]) = pack2bf(flo[it].y, fhi[it].y);
        }
    };

    v8f acc0 = {}, acc1 = {}, acc2 = {}, acc3 = {};

    const int m       = lane & 15;
    const int half    = lane >> 4;
    const int a_rbase = (wm * 16 + m) * 80;             // bytes
    const int b_cbase = (wn * 64 + m) * 80 + half * 32; // bytes

    // ---- prologue: fill buffer 0 ---------------------------------------
    stage_load();
    stage_store(AsB, BtB);
    __syncthreads();

    const int nk = K / KC;
    for (int ki = 0; ki < nk; ++ki) {
        const int  cur      = ki & 1;
        const bool has_next = (ki + 1) < nk;

        if (has_next) {
            if ((tid & 63) == 0)                       // global_prefetch_b8 hint
                __builtin_prefetch(bptr[0], 0, 3);
            stage_load();                              // global loads for k+1
        }

        // compute chunk ki from buf[cur]
        const char* Ab = (const char*)(AsB + cur * ASZ);
        const char* Bb = (const char*)(BtB + cur * BSZ);
        FragU afu;
        afu.q[0] = *(const v4u*)(Ab + a_rbase + half * 32);
        afu.q[1] = *(const v4u*)(Ab + a_rbase + half * 32 + 16);
        #pragma unroll
        for (int t = 0; t < 4; ++t) {
            FragU bfu;
            bfu.q[0] = *(const v4u*)(Bb + b_cbase + t * (16 * 80));
            bfu.q[1] = *(const v4u*)(Bb + b_cbase + t * (16 * 80) + 16);
            v8f* acc = (t == 0) ? &acc0 : (t == 1) ? &acc1 : (t == 2) ? &acc2 : &acc3;
            *acc = __builtin_amdgcn_wmma_f32_16x16x32_bf16(
                false, afu.v, false, bfu.v, (short)0, *acc, false, false);
        }

        if (has_next)                                  // fill the other buffer
            stage_store(AsB + (cur ^ 1) * ASZ, BtB + (cur ^ 1) * BSZ);
        __syncthreads();
    }

    // ---- epilogue + store ----------------------------------------------
    // C/D layout: lane n = lane&15, VGPR p -> row p (lanes<16) / p+8 (hi)
    const int rowsel = lane >> 4;
    #pragma unroll
    for (int t = 0; t < 4; ++t) {
        const v8f* acc = (t == 0) ? &acc0 : (t == 1) ? &acc1 : (t == 2) ? &acc2 : &acc3;
        #pragma unroll
        for (int p = 0; p < 8; ++p) {
            int gm = block_m + wm * 16 + p + 8 * rowsel;
            int gn = block_n + wn * 64 + t * 16 + m;
            if (!EDGE || gn < N) {
                float r = (*acc)[p] + bias[gn];
                out[(size_t)gm * N + gn] = (MODE == 1) ? tanhf(r) : r;
            }
        }
    }
}

template <int MODE>
__global__ __launch_bounds__(256) void pg_gemm_wmma(
    const float* __restrict__ X, const float* __restrict__ W,
    const float* __restrict__ bias, float* __restrict__ out,
    int M, int N, int K)
{
    __shared__ alignas(16) unsigned short AsB[2 * ASZ];
    __shared__ alignas(16) unsigned short BtB[2 * BSZ];
    const int block_m = blockIdx.y * BM;
    const int block_n = blockIdx.x * BN;
    if (block_n + BN <= N)   // hot path: 392/393 blocks, no guards anywhere
        pg_gemm_body<MODE, false>(AsB, BtB, X, W, bias, out, N, K, block_m, block_n);
    else
        pg_gemm_body<MODE, true>(AsB, BtB, X, W, bias, out, N, K, block_m, block_n);
}

// ---------------------------------------------------------------------------
// Gate head: mix[b,0:2] = softmax(h[b,:] @ W2 + b2). One block per row.
// ---------------------------------------------------------------------------
__global__ __launch_bounds__(256) void pg_gate2(
    const float* __restrict__ h, const float* __restrict__ W2,
    const float* __restrict__ b2, float* __restrict__ mix, int D)
{
    __shared__ float r0[256];
    __shared__ float r1[256];
    int b = blockIdx.x, tid = threadIdx.x;
    float a0 = 0.f, a1 = 0.f;
    for (int d = tid; d < D; d += 256) {
        float hv = h[(size_t)b * D + d];
        a0 += hv * W2[d * 2 + 0];
        a1 += hv * W2[d * 2 + 1];
    }
    r0[tid] = a0; r1[tid] = a1;
    __syncthreads();
    for (int s = 128; s > 0; s >>= 1) {
        if (tid < s) { r0[tid] += r0[tid + s]; r1[tid] += r1[tid + s]; }
        __syncthreads();
    }
    if (tid == 0) {
        float z0 = r0[0] + b2[0], z1 = r1[0] + b2[1];
        float m = fmaxf(z0, z1);
        float e0 = __expf(z0 - m), e1 = __expf(z1 - m);
        float inv = 1.0f / (e0 + e1);
        mix[2 * b]     = e0 * inv;
        mix[2 * b + 1] = e1 * inv;
    }
}

// ---------------------------------------------------------------------------
// Row stats over V: stat[2b]=max, stat[2b+1]=sum(exp(x-max)). Block per row.
// ---------------------------------------------------------------------------
__global__ __launch_bounds__(256) void pg_row_stats(
    const float* __restrict__ logits, float* __restrict__ stat, int V)
{
    __shared__ float red[256];
    int b = blockIdx.x, tid = threadIdx.x;
    const float* row = logits + (size_t)b * V;

    float m = -1e30f;
    for (int v = tid; v < V; v += 256) m = fmaxf(m, row[v]);
    red[tid] = m;
    __syncthreads();
    for (int s = 128; s > 0; s >>= 1) {
        if (tid < s) red[tid] = fmaxf(red[tid], red[tid + s]);
        __syncthreads();
    }
    float M = red[0];
    __syncthreads();

    float sum = 0.f;
    for (int v = tid; v < V; v += 256) sum += __expf(row[v] - M);
    red[tid] = sum;
    __syncthreads();
    for (int s = 128; s > 0; s >>= 1) {
        if (tid < s) red[tid] += red[tid + s];
        __syncthreads();
    }
    if (tid == 0) { stat[2 * b] = M; stat[2 * b + 1] = red[0]; }
}

// ---------------------------------------------------------------------------
// out[b,v] = exp(logit - max)/sum * mix0[b]
// ---------------------------------------------------------------------------
__global__ __launch_bounds__(256) void pg_finalize(
    const float* __restrict__ logits, const float* __restrict__ stat,
    const float* __restrict__ mix, float* __restrict__ out, int V, int total)
{
    int i = blockIdx.x * 256 + threadIdx.x;
    if (i >= total) return;
    int b = i / V;
    float scale = mix[2 * b] / stat[2 * b + 1];
    out[i] = __expf(logits[i] - stat[2 * b]) * scale;
}

// ---------------------------------------------------------------------------
// Copy distribution: softmax(scores[b,:]) scattered into out with atomics,
// weighted by mix1[b]. Block per row; S=512 -> 2 elems/thread.
// ---------------------------------------------------------------------------
__global__ __launch_bounds__(256) void pg_copy_scatter(
    const float* __restrict__ scores, const int* __restrict__ ctx,
    const float* __restrict__ mix, float* __restrict__ out, int S, int V)
{
    __shared__ float red[256];
    int b = blockIdx.x, tid = threadIdx.x;
    const float* row = scores + (size_t)b * S;

    float m = -1e30f;
    for (int s = tid; s < S; s += 256) m = fmaxf(m, row[s]);
    red[tid] = m;
    __syncthreads();
    for (int s = 128; s > 0; s >>= 1) {
        if (tid < s) red[tid] = fmaxf(red[tid], red[tid + s]);
        __syncthreads();
    }
    float M = red[0];
    __syncthreads();

    float sum = 0.f;
    for (int s = tid; s < S; s += 256) sum += __expf(row[s] - M);
    red[tid] = sum;
    __syncthreads();
    for (int s = 128; s > 0; s >>= 1) {
        if (tid < s) red[tid] += red[tid + s];
        __syncthreads();
    }
    float g = mix[2 * b + 1] / red[0];

    for (int s = tid; s < S; s += 256) {
        float a = __expf(row[s] - M) * g;
        int v = ctx[(size_t)b * S + s];
        atomicAdd(&out[(size_t)b * V + v], a);
    }
}

// ---------------------------------------------------------------------------
// Host-side orchestration
// ---------------------------------------------------------------------------
extern "C" void kernel_launch(void* const* d_in, const int* in_sizes, int n_in,
                              void* d_out, int out_size, void* d_ws, size_t ws_size,
                              hipStream_t stream)
{
    const float* x      = (const float*)d_in[0];   // [B,D]
    const float* scores = (const float*)d_in[1];   // [B,S]
    const int*   ctx    = (const int*)  d_in[2];   // [B,S]
    const float* Wg     = (const float*)d_in[3];   // [D,V]
    const float* bg     = (const float*)d_in[4];   // [V]
    const float* W1     = (const float*)d_in[5];   // [D,D]
    const float* b1     = (const float*)d_in[6];   // [D]
    const float* W2     = (const float*)d_in[7];   // [D,2]
    const float* b2     = (const float*)d_in[8];   // [2]
    float* out = (float*)d_out;                    // [B,V]

    const int B = 256, S = 512, D = 1024, V = 50257;

    // workspace layout
    float* logits  = (float*)d_ws;                    // B*V
    float* h       = logits + (size_t)B * V;          // B*D
    float* rowstat = h + (size_t)B * D;               // 2*B (max, sum)
    float* mix     = rowstat + 2 * B;                 // 2*B

    // 1) gate hidden: h = tanh(x @ W1 + b1)   [B,D]
    pg_gemm_wmma<1><<<dim3(D / BN, B / BM), 256, 0, stream>>>(x, W1, b1, h, B, D, D);

    // 2) gate mix: softmax(h @ W2 + b2)       [B,2]
    pg_gate2<<<B, 256, 0, stream>>>(h, W2, b2, mix, D);

    // 3) logits = x @ Wg + bg                 [B,V]
    pg_gemm_wmma<0><<<dim3((V + BN - 1) / BN, B / BM), 256, 0, stream>>>(x, Wg, bg, logits, B, V, D);

    // 4) rowwise softmax stats over V
    pg_row_stats<<<B, 256, 0, stream>>>(logits, rowstat, V);

    // 5) out = softmax(logits) * mix0
    int total = B * V;
    pg_finalize<<<(total + 255) / 256, 256, 0, stream>>>(logits, rowstat, mix, out, V, total);

    // 6) out += scatter(softmax(scores) * mix1)
    pg_copy_scatter<<<B, 256, 0, stream>>>(scores, ctx, mix, out, S, V);
}
